// GAU_48627619726042
// MI455X (gfx1250) — matrix operand order repaired and checked by reading, not compiled, not run on hardware
//
#include <hip/hip_runtime.h>
#include <hip/hip_bf16.h>

typedef __attribute__((ext_vector_type(16))) __bf16 v16bf;
typedef __attribute__((ext_vector_type(8)))  float  v8f;
typedef __attribute__((ext_vector_type(4)))  float  f32x4;

#define B_ 8
#define T_ 2048
#define D_ 512
#define E_ 1024
#define S_ 128

union Frag { v16bf v; f32x4 q[2]; };

// async 16B copy: global -> LDS, tracked by ASYNCcnt (CDNA5 async data mover path)
#define ASYNC_CP16(ldsOff, gp)                                           \
    asm volatile("global_load_async_to_lds_b128 %0, %1, off"             \
                 :: "v"(ldsOff), "v"(gp) : "memory")
#define ASYNC_WAIT0() asm volatile("s_wait_asynccnt 0x0" ::: "memory")

// -----------------------------------------------------------------------------
// 256x128-tile bf16 GEMM, double-buffered LDS fed by async global->LDS copies:
//   C(M,N) = A(MxK) * Bn(NxK)^T, row-major bf16, lda = ldb = K.
// 8 waves per block in a 4x2 grid; each wave computes a 64x64 subtile
// (16 accumulators) with v_wmma_f32_16x16x32_bf16.  One barrier per K-step.
// Mode-specific fused epilogues.
// -----------------------------------------------------------------------------
template <int MODE>
__global__ __launch_bounds__(256)
void gemm_bf16(const __bf16* __restrict__ A, const __bf16* __restrict__ Bn,
               size_t aBS, size_t bBS, size_t oBS, int K,
               float* __restrict__ outF, __bf16* __restrict__ outB,
               __bf16* __restrict__ outB2, float* __restrict__ auxF,
               const float* __restrict__ wrel, const __bf16* __restrict__ uG,
               const float* __restrict__ xRes)
{
    __shared__ __bf16 At[2][256][32];   // 32 KB
    __shared__ __bf16 Bt[2][128][32];   // 16 KB

    const int z = blockIdx.z;
    const __bf16* Ab = A  + (size_t)z * aBS;
    const __bf16* Bb = Bn + (size_t)z * bBS;

    const int blockRow = blockIdx.y * 256;
    const int blockCol = blockIdx.x * 128;

    const int tid  = threadIdx.x;
    const int lane = tid & 31;
    const int wave = tid >> 5;
    const int waveRow = (wave >> 1) * 64;   // 4 wave-rows
    const int waveCol = (wave & 1) * 64;    // 2 wave-cols
    const int h   = lane >> 4;              // lane half
    const int l16 = lane & 15;

    // cooperative tile staging: 256 threads; A = 256x32, B = 128x32 bf16
    const int ldRow = tid >> 1;             // 0..127
    const int ldCol = (tid & 1) * 16;       // 0 or 16 (16 bf16 = 32B)

    const __bf16* agBase = Ab + (size_t)(blockRow + ldRow) * K + ldCol;
    const __bf16* ag2    = agBase + (size_t)128 * K;          // rows 128..255
    const __bf16* bgBase = Bb + (size_t)(blockCol + ldRow) * K + ldCol;

    // LDS byte offsets for this thread's 6 async-copy destinations, per buffer
    // (generic shared-pointer truncated to 32 bits == LDS offset, ISA 10.2)
    unsigned oA0[2], oA2[2], oB0[2];
#pragma unroll
    for (int b = 0; b < 2; ++b) {
        oA0[b] = (unsigned)(uintptr_t)&At[b][ldRow][ldCol];
        oA2[b] = (unsigned)(uintptr_t)&At[b][ldRow + 128][ldCol];
        oB0[b] = (unsigned)(uintptr_t)&Bt[b][ldRow][ldCol];
    }

    v8f acc[4][4] = {};

    // ---- prologue: async-fill buffer 0 with K-step 0 ----
    ASYNC_CP16(oA0[0],      agBase);
    ASYNC_CP16(oA0[0] + 16u, agBase + 8);
    ASYNC_CP16(oA2[0],      ag2);
    ASYNC_CP16(oA2[0] + 16u, ag2 + 8);
    ASYNC_CP16(oB0[0],      bgBase);
    ASYNC_CP16(oB0[0] + 16u, bgBase + 8);
    ASYNC_WAIT0();
    __syncthreads();

    int buf = 0;
    for (int k0 = 0; k0 < K; k0 += 32) {
        const bool hasNext = (k0 + 32) < K;

        // kick off async copies for the next K-step into the other buffer;
        // they land in LDS while the WMMAs below execute
        if (hasNext) {
            const int kn = k0 + 32;
            const int nb = buf ^ 1;
            ASYNC_CP16(oA0[nb],       agBase + kn);
            ASYNC_CP16(oA0[nb] + 16u, agBase + kn + 8);
            ASYNC_CP16(oA2[nb],       ag2 + kn);
            ASYNC_CP16(oA2[nb] + 16u, ag2 + kn + 8);
            ASYNC_CP16(oB0[nb],       bgBase + kn);
            ASYNC_CP16(oB0[nb] + 16u, bgBase + kn + 8);
        }

        // build all fragments, then issue WMMAs back-to-back
        const __bf16 (*Ac)[32] = At[buf];
        const __bf16 (*Bc)[32] = Bt[buf];
        Frag af[4], bfr[4];
#pragma unroll
        for (int i = 0; i < 4; ++i) {
            int m = waveRow + 16 * i + l16;
            af[i].q[0] = *(const f32x4*)&Ac[m][8 * h];
            af[i].q[1] = *(const f32x4*)&Ac[m][8 * h + 16];
        }
#pragma unroll
        for (int j = 0; j < 4; ++j) {
            int n = waveCol + 16 * j + l16;
            bfr[j].q[0] = *(const f32x4*)&Bc[n][8 * h];
            bfr[j].q[1] = *(const f32x4*)&Bc[n][8 * h + 16];
        }
#pragma unroll
        for (int i = 0; i < 4; ++i)
#pragma unroll
        for (int j = 0; j < 4; ++j) {
            acc[i][j] = __builtin_amdgcn_wmma_f32_16x16x32_bf16(
                false, af[i].v, false, bfr[j].v, (short)0, acc[i][j],
                false, false);
        }

        if (hasNext) ASYNC_WAIT0();   // our async copies done before barrier
        __syncthreads();
        buf ^= 1;
    }

    // epilogue: C element (r,lane): m = r + 8h, n = l16
#pragma unroll
    for (int i = 0; i < 4; ++i)
#pragma unroll
    for (int j = 0; j < 4; ++j)
#pragma unroll
    for (int r = 0; r < 8; ++r) {
        int m = blockRow + waveRow + 16 * i + r + 8 * h;
        int n = blockCol + waveCol + 16 * j + l16;
        float val = acc[i][j][r];
        if (MODE == 0) {
            // silu, then split uv -> u (bf16), v^T (bf16, N-major), base (f32)
            val = val / (1.0f + __expf(-val));
            if (n < E_) {
                outB[(size_t)m * E_ + n] = (__bf16)val;
            } else if (n < 2 * E_) {
                int e = n - E_; int b = m >> 11; int t = m & (T_ - 1);
                outB2[((size_t)b * E_ + e) * T_ + t] = (__bf16)val;
            } else {
                auxF[(size_t)m * S_ + (n - 2 * E_)] = val;
            }
        } else if (MODE == 1) {
            // qk: relative bias, relu(.)^2 / S, store bf16 scores
            float bias = wrel[T_ - 1 + n - m];
            val = (val + bias) * 0.08838834764831845f;   // 1/sqrt(128)
            val = fmaxf(val, 0.0f);
            val = val * val;
            (outB + (size_t)z * oBS)[(size_t)m * T_ + n] = (__bf16)val;
        } else if (MODE == 2) {
            // gate by u
            float u = (float)(uG + (size_t)z * oBS)[(size_t)m * E_ + n];
            (outB + (size_t)z * oBS)[(size_t)m * E_ + n] = (__bf16)(val * u);
        } else {
            // final projection + residual
            val += xRes[(size_t)m * D_ + n];
            outF[(size_t)m * D_ + n] = val;
        }
    }
}

// -----------------------------------------------------------------------------
__global__ void f32_to_bf16(const float* __restrict__ src, __bf16* __restrict__ dst, int n)
{
    int i = blockIdx.x * 256 + threadIdx.x;
    if (i < n) dst[i] = (__bf16)src[i];
}

// one wave (32 lanes) per row of 512; 8 rows per 256-thread block
__global__ __launch_bounds__(256)
void norm_kernel(const float* __restrict__ x, const float* __restrict__ g,
                 __bf16* __restrict__ xn)
{
    int row  = blockIdx.x * 8 + (threadIdx.x >> 5);
    int lane = threadIdx.x & 31;
    const float* xr = x + (size_t)row * D_;
    float vals[16];
    float s = 0.0f;
#pragma unroll
    for (int i = 0; i < 16; ++i) {
        float v = xr[lane + 32 * i];
        vals[i] = v;
        s += v * v;
    }
#pragma unroll
    for (int m = 16; m > 0; m >>= 1) s += __shfl_xor(s, m, 32);
    float norm  = sqrtf(s) * 0.04419417382415922f;     // * D^-0.5
    float scale = g[0] / fmaxf(norm, 1e-5f);
#pragma unroll
    for (int i = 0; i < 16; ++i)
        xn[(size_t)row * D_ + lane + 32 * i] = (__bf16)(vals[i] * scale);
}

// gamma/beta affine + RoPE -> q,k in bf16; one block per (b,t) row, 64 threads
__global__ void rope_kernel(const float* __restrict__ baseF,
                            const float* __restrict__ gamma,
                            const float* __restrict__ beta,
                            __bf16* __restrict__ qB, __bf16* __restrict__ kB)
{
    int m = blockIdx.x;
    int s = threadIdx.x;                   // 0..63
    int t = m & (T_ - 1);
    float f   = powf(10000.0f, (float)s * (1.0f / 64.0f));
    float ang = (float)t * f;
    float sv  = sinf(ang), cv = cosf(ang);
    float b1 = baseF[(size_t)m * S_ + s];
    float b2 = baseF[(size_t)m * S_ + s + 64];
#pragma unroll
    for (int i = 0; i < 2; ++i) {
        float x1 = b1 * gamma[i * S_ + s]      + beta[i * S_ + s];
        float x2 = b2 * gamma[i * S_ + s + 64] + beta[i * S_ + s + 64];
        __bf16* dst = i ? kB : qB;
        dst[(size_t)m * S_ + s]      = (__bf16)(x1 * cv - x2 * sv);
        dst[(size_t)m * S_ + s + 64] = (__bf16)(x2 * cv + x1 * sv);
    }
}

// -----------------------------------------------------------------------------
extern "C" void kernel_launch(void* const* d_in, const int* in_sizes, int n_in,
                              void* d_out, int out_size, void* d_ws, size_t ws_size,
                              hipStream_t stream)
{
    (void)in_sizes; (void)n_in; (void)out_size; (void)ws_size;
    const float* x     = (const float*)d_in[0];
    const float* g     = (const float*)d_in[1];
    const float* w_uv  = (const float*)d_in[2];
    const float* gamma = (const float*)d_in[3];
    const float* beta  = (const float*)d_in[4];
    const float* wrel  = (const float*)d_in[5];
    const float* w_o   = (const float*)d_in[6];
    float* out = (float*)d_out;

    char*  ws  = (char*)d_ws;
    size_t off = 0;
    auto take = [&](size_t bytes) -> char* {
        char* p = ws + off;
        off = (off + bytes + 255) & ~(size_t)255;
        return p;
    };

    __bf16* wuvB  = (__bf16*)take((size_t)(2 * E_ + S_) * D_ * 2); // 2.2 MB
    __bf16* woB   = (__bf16*)take((size_t)D_ * E_ * 2);            // 1.0 MB
    __bf16* xnB   = (__bf16*)take((size_t)B_ * T_ * D_ * 2);       // 16.8 MB
    __bf16* uB    = (__bf16*)take((size_t)B_ * T_ * E_ * 2);       // 33.6 MB
    __bf16* vTB   = (__bf16*)take((size_t)B_ * E_ * T_ * 2);       // 33.6 MB (N-major)
    float*  baseF = (float*) take((size_t)B_ * T_ * S_ * 4);       // 8.4 MB
    __bf16* qB    = (__bf16*)take((size_t)B_ * T_ * S_ * 2);       // 4.2 MB
    __bf16* kB    = (__bf16*)take((size_t)B_ * T_ * S_ * 2);       // 4.2 MB
    __bf16* scB   = (__bf16*)take((size_t)B_ * T_ * T_ * 2);       // 67 MB
    __bf16* gateB = (__bf16*)take((size_t)B_ * T_ * E_ * 2);       // 33.6 MB

    // weights -> bf16
    { int n = (2 * E_ + S_) * D_; f32_to_bf16<<<(n + 255) / 256, 256, 0, stream>>>(w_uv, wuvB, n); }
    { int n = D_ * E_;            f32_to_bf16<<<(n + 255) / 256, 256, 0, stream>>>(w_o, woB, n); }

    // xn = normalize(x) * g   (bf16)
    norm_kernel<<<(B_ * T_) / 8, 256, 0, stream>>>(x, g, xnB);

    // GEMM1: uv = silu(xn @ w_uv^T); split into u, v^T, base
    gemm_bf16<0><<<dim3((2 * E_ + S_) / 128, (B_ * T_) / 256, 1), 256, 0, stream>>>(
        xnB, wuvB, 0, 0, 0, D_,
        nullptr, uB, vTB, baseF, nullptr, nullptr, nullptr);

    // gamma/beta + RoPE -> q,k bf16
    rope_kernel<<<B_ * T_, 64, 0, stream>>>(baseF, gamma, beta, qB, kB);

    // scores = relu((q k^T + w_rel)/sqrt(S))^2   (per batch)
    gemm_bf16<1><<<dim3(T_ / 128, T_ / 256, B_), 256, 0, stream>>>(
        qB, kB, (size_t)T_ * S_, (size_t)T_ * S_, (size_t)T_ * T_, S_,
        nullptr, scB, nullptr, nullptr, wrel, nullptr, nullptr);

    // gated = u * (scores @ v)   (per batch; v^T is N-major)
    gemm_bf16<2><<<dim3(E_ / 128, T_ / 256, B_), 256, 0, stream>>>(
        scB, vTB, (size_t)T_ * T_, (size_t)E_ * T_, (size_t)T_ * E_, T_,
        nullptr, gateB, nullptr, nullptr, nullptr, uB, nullptr);

    // out = gated @ w_o^T + x
    gemm_bf16<3><<<dim3(D_ / 128, (B_ * T_) / 256, 1), 256, 0, stream>>>(
        gateB, woB, 0, 0, 0, E_,
        out, nullptr, nullptr, nullptr, nullptr, nullptr, x);
}
